// HyperResBlock1d_14431090114938
// MI455X (gfx1250) — compile-verified, matrix operand order checked by reading, not run on hardware
//
#include <hip/hip_runtime.h>
#include <hip/hip_bf16.h>

// Problem constants (match reference)
#define B_  16
#define C_  256      // IN_N == MID_N == OUT_N
#define L_  4096
#define WH_ 1024
#define NW_ 768      // C_*K (K=3)
#define EPS_ 1e-5f

typedef __attribute__((ext_vector_type(16))) __bf16 v16bf;
typedef __attribute__((ext_vector_type(8)))  __bf16 v8bf;
typedef __attribute__((ext_vector_type(8)))  float  v8f;
typedef __attribute__((ext_vector_type(4)))  unsigned v4u;
typedef __attribute__((ext_vector_type(8)))  unsigned v8u;

union U16 { v16bf v; v8bf h[2]; unsigned int w[8]; };

__device__ __forceinline__ v8f wmma_bf16(const U16& a, const U16& b, v8f c) {
  // emits v_wmma_f32_16x16x32_bf16
  return __builtin_amdgcn_wmma_f32_16x16x32_bf16(false, a.v, false, b.v, (short)0, c,
                                                 false, false);
}

// ---------------- conversion kernels ----------------

__global__ void cvt_f32_bf16(const float* __restrict__ in, __bf16* __restrict__ out, int n) {
  int i = blockIdx.x * 256 + threadIdx.x;
  if (i < n) out[i] = (__bf16)in[i];
}

// x [B,C,L] f32 -> xT [B,L,C] bf16 (transposed so conv B-operand rows are contiguous)
__global__ void cvt_xT(const float* __restrict__ x, __bf16* __restrict__ xT) {
  unsigned i = blockIdx.x * 256u + threadIdx.x;   // c fastest
  unsigned c = i & (C_ - 1);
  unsigned rest = i >> 8;
  unsigned l = rest & (L_ - 1);
  unsigned b = rest >> 12;
  xT[i] = (__bf16)x[((size_t)(b * C_ + c)) * L_ + l];
}

// ---------------- hypernet weight generation: Out[m,n] = att[m,:] . w[n,:] ----------------
// A: [B*C, WH] bf16 row-major, Bw: [NW, WH] bf16 row-major (= B^T, same load pattern as A).
// Each wave computes a 16(M) x 64(N) strip: one A fetch feeds 4 WMMAs; all 4 B sub-tiles
// are fetched before the WMMA burst so loads clause under a single wait.
__global__ void hyper_gemm(const __bf16* __restrict__ A, const __bf16* __restrict__ Bw,
                           const float* __restrict__ bias, __bf16* __restrict__ Wout) {
  int tid = threadIdx.x;
  int lane = tid & 31, wv = tid >> 5;     // 4 waves/block, each owns one M-tile
  int r = lane & 15, hi = lane >> 4;
  int nb0 = blockIdx.x * 4;               // 4 consecutive N-tiles per wave
  int mb  = blockIdx.y * 4 + wv;          // 0..255
  const __bf16* arow = A + (size_t)(mb * 16 + r) * WH_;
  v8f acc[4] = {};
  for (int k0 = 0; k0 < WH_; k0 += 32) {
    if (k0 + 256 < WH_) __builtin_prefetch(arow + k0 + 256, 0, 3);   // global_prefetch_b8
    U16 a;
    a.h[0] = *(const v8bf*)(arow + k0 + hi * 8);
    a.h[1] = *(const v8bf*)(arow + k0 + 16 + hi * 8);
    U16 bx[4];
#pragma unroll
    for (int s = 0; s < 4; ++s) {
      const __bf16* brow = Bw + (size_t)((nb0 + s) * 16 + r) * WH_;
      bx[s].h[0] = *(const v8bf*)(brow + k0 + hi * 8);
      bx[s].h[1] = *(const v8bf*)(brow + k0 + 16 + hi * 8);
    }
#pragma unroll
    for (int s = 0; s < 4; ++s)
      acc[s] = wmma_bf16(a, bx[s], acc[s]);   // 4 independent WMMAs back-to-back
  }
  // C/D layout: col = lane&15, row = i + hi*8
  int m0 = mb * 16 + hi * 8;
#pragma unroll
  for (int s = 0; s < 4; ++s) {
    int n  = (nb0 + s) * 16 + r;
    int cc = n / 3, t = n - 3 * cc;       // reshape [..,768] -> [..,C,3]
    float bv = bias[n];
#pragma unroll
    for (int i = 0; i < 8; ++i) {
      int m = m0 + i;
      int bb = m >> 8, o = m & (C_ - 1);
      Wout[(((size_t)(bb * 3 + t) * C_) + o) * C_ + cc] = (__bf16)(acc[s][i] + bv);
    }
  }
}

// b1[b,o] = att1[b,o,:] . b1_w[0,:] + b1_b   (one wave per row)
__global__ void bias_gemv(const float* __restrict__ att, const float* __restrict__ bw,
                          const float* __restrict__ bb, float* __restrict__ out) {
  int row  = blockIdx.x * 8 + (threadIdx.x >> 5);
  int lane = threadIdx.x & 31;
  const float* a = att + (size_t)row * WH_;
  float s = 0.f;
  for (int k = lane; k < WH_; k += 32) s += a[k] * bw[k];
  for (int m = 16; m; m >>= 1) s += __shfl_xor(s, m, 32);
  if (lane == 0) out[row] = s + bb[0];
}

// ---------------- per-sample conv1d as WMMA GEMM ----------------
// Block = 128 threads (4 waves). The 3x16x256 weight A-panel for (b, o-tile) is
// staged into LDS with ONE Tensor-Data-Mover descriptor: 2D tile (4096 elems x 3 taps),
// tensor_dim0_stride = 64K elems (128KB tap stride), TDM pad feature inserts 16B after
// every 512B row -> 528B LDS row stride (bank rotation of 4 DWORD-banks per row).
// Wave 0 issues it, drains TENSORcnt, then the block barriers; all 4 waves then feed
// WMMA A-operands from LDS while streaming x from L2.
// Each wave computes a 16(o) x 64(l) strip: one LDS A fetch feeds 4 WMMAs.
// MODE 1: +bias, relu, store bf16 transposed [b,l,o]
// MODE 2: raw store bf16 transposed + BN stats (sum, sumsq) via f32 atomics
// MODE 3: raw store f32 [b,o,l] + BN stats
template<int MODE>
__global__ void conv_wmma(const __bf16* __restrict__ Wt,   // [b][t][o][c] bf16
                          const __bf16* __restrict__ xT,   // [b][l][c]  bf16
                          const float*  __restrict__ bias, // [b][o] (MODE 1)
                          __bf16*       __restrict__ outT, // [b][l][o] (MODE 1,2)
                          float*        __restrict__ outF, // [b][o][l] (MODE 3)
                          float*        __restrict__ stats)// [2][C] (MODE 2,3)
{
  constexpr int LDSP = 264;  // 528B padded row stride, produced by TDM pad feature
  __shared__ __align__(16) __bf16 As[3 * 16 * LDSP];
  int tid  = threadIdx.x;
  int lane = tid & 31, wv = tid >> 5;
  int r = lane & 15, hi = lane >> 4;
  int lt = blockIdx.x;                    // block owns l-range [lt*256, lt*256+256)
  int ot = blockIdx.y;                    // 0..15
  int b  = blockIdx.z;                    // 0..15

  // ---- TDM: stage whole weight panel (3 x 16 x 256 bf16) with one descriptor ----
  if (wv == 0) {
    unsigned lds0 = (unsigned)(unsigned long long)(void*)As;     // LDS byte offset
    unsigned long long ga =
        (unsigned long long)(Wt + ((size_t)b * 3 * C_ + ot * 16) * C_);
    v4u g0;
    g0[0] = 1u;                                   // count=1 valid descriptor
    g0[1] = lds0;                                 // lds_addr (bytes)
    g0[2] = (unsigned)ga;                         // global_addr[31:0]
    g0[3] = ((unsigned)(ga >> 32) & 0x01FFFFFFu)  // global_addr[56:32]
            | 0x80000000u;                        // type=2 ("image")
    v8u g1;
    g1[0] = (1u << 16)      // data_size = 2 bytes
          | (1u << 20)      // pad_enable
          | (6u << 22)      // pad_interval: every 128 DWORDs (512B row)
          | (3u << 25);     // pad_amount: 4 DWORDs (16B)
    g1[1] = (4096u & 0xFFFFu) << 16;              // tensor_dim0[15:0] (=4096 elems)
    g1[2] = (4096u >> 16) | (3u << 16);           // tensor_dim0[31:16] | tensor_dim1=3
    g1[3] = (4096u << 16);                        // tile_dim0 = 4096 elems (16 rows)
    g1[4] = 3u;                                   // tile_dim1 = 3 (taps), tile_dim2=0
    g1[5] = 65536u;                               // tensor_dim0_stride = C*C elems
    g1[6] = 0u;
    g1[7] = 0u;
    asm volatile("tensor_load_to_lds %0, %1" :: "s"(g0), "s"(g1) : "memory");
    __builtin_amdgcn_s_wait_tensorcnt(0);
  }
  __syncthreads();

  // ---- main WMMA loop ----
  v8f acc[4] = {};
  const int lbase = lt * 256 + wv * 64;
#pragma unroll
  for (int t = 0; t < 3; ++t) {
    const __bf16* arow = As + (t * 16 + r) * LDSP;
#pragma unroll 2
    for (int k0 = 0; k0 < C_; k0 += 32) {
      U16 a;                               // ds_load_b128 x2 from staged panel
      a.h[0] = *(const v8bf*)(arow + k0 + hi * 8);
      a.h[1] = *(const v8bf*)(arow + k0 + 16 + hi * 8);
      U16 bx[4];
#pragma unroll
      for (int s = 0; s < 4; ++s) {
        int lsrc = lbase + s * 16 + r + t - 1;          // padding='same'
        bool ok  = (unsigned)lsrc < (unsigned)L_;
        if (ok) {
          const __bf16* xrow = xT + ((size_t)b * L_ + lsrc) * C_;
          bx[s].h[0] = *(const v8bf*)(xrow + k0 + hi * 8);
          bx[s].h[1] = *(const v8bf*)(xrow + k0 + 16 + hi * 8);
        } else {
#pragma unroll
          for (int j = 0; j < 8; ++j) bx[s].w[j] = 0u;  // zero rows outside [0,L)
        }
      }
#pragma unroll
      for (int s = 0; s < 4; ++s)
        acc[s] = wmma_bf16(a, bx[s], acc[s]);           // EXEC all-ones here
    }
  }

  // ---- epilogue ----
#pragma unroll
  for (int s = 0; s < 4; ++s) {
    int l = lbase + s * 16 + r;
#pragma unroll
    for (int i = 0; i < 8; ++i) {
      int o = ot * 16 + hi * 8 + i;
      float v = acc[s][i];
      if (MODE == 1) {
        v = fmaxf(v + bias[b * C_ + o], 0.f);
        outT[((size_t)b * L_ + l) * C_ + o] = (__bf16)v;
      } else if (MODE == 2) {
        outT[((size_t)b * L_ + l) * C_ + o] = (__bf16)v;
      } else {
        outF[((size_t)(b * C_ + o)) * L_ + l] = v;
      }
      if (MODE >= 2) {
        float sm = v, s2 = v * v;           // reduce the 16 lanes holding this row
        sm += __shfl_xor(sm, 1, 32);  s2 += __shfl_xor(s2, 1, 32);
        sm += __shfl_xor(sm, 2, 32);  s2 += __shfl_xor(s2, 2, 32);
        sm += __shfl_xor(sm, 4, 32);  s2 += __shfl_xor(s2, 4, 32);
        sm += __shfl_xor(sm, 8, 32);  s2 += __shfl_xor(s2, 8, 32);
        if (r == 0) {
          atomicAdd(&stats[o], sm);         // global_atomic_add_f32
          atomicAdd(&stats[C_ + o], s2);
        }
      }
    }
  }
}

// ---------------- BN finalize / apply ----------------

__global__ void bn_finalize(const float* __restrict__ stats, const float* __restrict__ w,
                            const float* __restrict__ bb, float* __restrict__ scale,
                            float* __restrict__ shift) {
  int c = threadIdx.x;
  const float inv_cnt = 1.f / (float)(B_ * L_);
  float mean = stats[c] * inv_cnt;
  float var  = stats[C_ + c] * inv_cnt - mean * mean;
  float sc   = w[c] * rsqrtf(var + EPS_);
  scale[c] = sc;
  shift[c] = bb[c] - mean * sc;
}

// in-place BN + relu on transposed bf16 activations [b,l,c]
__global__ void bn_relu_bf16(__bf16* __restrict__ hT, const float* __restrict__ scale,
                             const float* __restrict__ shift) {
  unsigned i = blockIdx.x * 256u + threadIdx.x;
  unsigned c = i & (C_ - 1);
  float v = (float)hT[i] * scale[c] + shift[c];
  hT[i] = (__bf16)fmaxf(v, 0.f);
}

// ---------------- SE module ----------------

// s[b,c] = mean_l relu(bn3(h3raw))   (one block per (b,c) row)
__global__ void se_mean(const float* __restrict__ h, const float* __restrict__ sc,
                        const float* __restrict__ sh, float* __restrict__ s) {
  __shared__ float red[256];
  int bc = blockIdx.x;
  int c  = bc & (C_ - 1);
  const float* row = h + (size_t)bc * L_;
  float a = 0.f, k1 = sc[c], k2 = sh[c];
  for (int l = threadIdx.x; l < L_; l += 256)
    a += fmaxf(row[l] * k1 + k2, 0.f);
  red[threadIdx.x] = a;
  __syncthreads();
  for (int st = 128; st > 0; st >>= 1) {
    if (threadIdx.x < st) red[threadIdx.x] += red[threadIdx.x + st];
    __syncthreads();
  }
  if (threadIdx.x == 0) s[bc] = red[0] * (1.f / L_);
}

// gate[b,o] = sigmoid(W2 . relu(W1 . s + b1) + b2)   (one block per batch)
// The SE vector is staged into LDS with GLOBAL_LOAD_ASYNC_TO_LDS_B32 (ASYNCcnt path).
__global__ void se_gemv(const float* __restrict__ s, const float* __restrict__ w1,
                        const float* __restrict__ b1, const float* __restrict__ w2,
                        const float* __restrict__ b2, float* __restrict__ gate) {
  __shared__ __align__(16) float sv[C_];
  __shared__ float s1[C_];
  int b = blockIdx.x, o = threadIdx.x;
  {
    unsigned la = (unsigned)(unsigned long long)(void*)sv + (unsigned)o * 4u;
    unsigned long long ga = (unsigned long long)(s + b * C_ + o);
    asm volatile("global_load_async_to_lds_b32 %0, %1, off"
                 :: "v"(la), "v"(ga) : "memory");
    asm volatile("s_wait_asynccnt 0" ::: "memory");
  }
  __syncthreads();
  float a1 = b1[o];
  for (int c = 0; c < C_; ++c) a1 += w1[o * C_ + c] * sv[c];
  s1[o] = fmaxf(a1, 0.f);
  __syncthreads();
  float a2 = b2[o];
  for (int c = 0; c < C_; ++c) a2 += w2[o * C_ + c] * s1[c];
  gate[b * C_ + o] = 1.f / (1.f + expf(-a2));
}

// out = relu( relu(bn3(h3)) * gate + x )   over [b,c,l], l fastest
__global__ void final_fuse(const float* __restrict__ h, const float* __restrict__ sc,
                           const float* __restrict__ sh, const float* __restrict__ gate,
                           const float* __restrict__ x, float* __restrict__ out) {
  unsigned i = blockIdx.x * 256u + threadIdx.x;
  unsigned rest = i >> 12;
  unsigned c = rest & (C_ - 1);
  unsigned b = rest >> 8;
  float v = fmaxf(h[i] * sc[c] + sh[c], 0.f);
  out[i] = fmaxf(v * gate[b * C_ + c] + x[i], 0.f);
}

// ---------------- launch ----------------

extern "C" void kernel_launch(void* const* d_in, const int* in_sizes, int n_in,
                              void* d_out, int out_size, void* d_ws, size_t ws_size,
                              hipStream_t stream) {
  const float* x      = (const float*)d_in[0];
  const float* att[3] = {(const float*)d_in[1], (const float*)d_in[2], (const float*)d_in[3]};
  const float* ww[3]  = {(const float*)d_in[4], (const float*)d_in[6], (const float*)d_in[8]};
  const float* wbv[3] = {(const float*)d_in[5], (const float*)d_in[7], (const float*)d_in[9]};
  const float* b1_w   = (const float*)d_in[10];
  const float* b1_b   = (const float*)d_in[11];
  const float* bn2_w  = (const float*)d_in[12];
  const float* bn2_b  = (const float*)d_in[13];
  const float* bn3_w  = (const float*)d_in[14];
  const float* bn3_b  = (const float*)d_in[15];
  const float* se1_w  = (const float*)d_in[16];
  const float* se1_b  = (const float*)d_in[17];
  const float* se2_w  = (const float*)d_in[18];
  const float* se2_b  = (const float*)d_in[19];
  float* out = (float*)d_out;

  char* base = (char*)d_ws;
  size_t off = 0;
  auto carve = [&](size_t bytes) -> char* {
    char* r = base + off;
    off += (bytes + 1023) & ~(size_t)1023;
    return r;
  };
  __bf16* attB  = (__bf16*)carve((size_t)B_ * C_ * WH_ * 2);        // 8.4 MB (reused x3)
  __bf16* wBB   = (__bf16*)carve((size_t)NW_ * WH_ * 2);            // 1.6 MB (reused x3)
  __bf16* Wgen  = (__bf16*)carve((size_t)3 * B_ * 3 * C_ * C_ * 2); // 18.9 MB
  float*  b1v   = (float*) carve((size_t)B_ * C_ * 4);
  __bf16* xT0   = (__bf16*)carve((size_t)B_ * L_ * C_ * 2);         // 33.5 MB
  __bf16* xT1   = (__bf16*)carve((size_t)B_ * L_ * C_ * 2);         // 33.5 MB
  float*  h3    = (float*) carve((size_t)B_ * C_ * L_ * 4);         // 67 MB
  float*  stats = (float*) carve(4 * C_ * 4);                       // bn2:[0,2C) bn3:[2C,4C)
  float*  sc2   = (float*) carve(C_ * 4);
  float*  sh2   = (float*) carve(C_ * 4);
  float*  sc3   = (float*) carve(C_ * 4);
  float*  sh3   = (float*) carve(C_ * 4);
  float*  sbuf  = (float*) carve((size_t)B_ * C_ * 4);
  float*  gate  = (float*) carve((size_t)B_ * C_ * 4);

  const size_t wstep = (size_t)B_ * 3 * C_ * C_;   // per-layer generated-weight elems
  const int NEL = B_ * L_ * C_;                    // 16.7M activation elements

  hipMemsetAsync(stats, 0, 4 * C_ * 4, stream);

  // x -> transposed bf16
  cvt_xT<<<dim3(NEL / 256), 256, 0, stream>>>(x, xT0);

  // hypernet weight generation (3 WMMA GEMMs, M=4096 N=768 K=1024)
  for (int layer = 0; layer < 3; ++layer) {
    cvt_f32_bf16<<<dim3((B_ * C_ * WH_) / 256), 256, 0, stream>>>(att[layer], attB, B_ * C_ * WH_);
    cvt_f32_bf16<<<dim3((NW_ * WH_) / 256), 256, 0, stream>>>(ww[layer], wBB, NW_ * WH_);
    hyper_gemm<<<dim3(NW_ / 64, (B_ * C_) / 64), 128, 0, stream>>>(
        attB, wBB, wbv[layer], Wgen + (size_t)layer * wstep);
  }
  bias_gemv<<<dim3((B_ * C_) / 8), 256, 0, stream>>>(att[0], b1_w, b1_b, b1v);

  // layer 1: conv + bias + relu  (x -> h1, transposed bf16)
  conv_wmma<1><<<dim3(L_ / 256, C_ / 16, B_), 128, 0, stream>>>(
      Wgen, xT0, b1v, xT1, nullptr, nullptr);

  // layer 2: conv (raw) + BN stats, then BN+relu in place
  conv_wmma<2><<<dim3(L_ / 256, C_ / 16, B_), 128, 0, stream>>>(
      Wgen + wstep, xT1, nullptr, xT0, nullptr, stats);
  bn_finalize<<<1, 256, 0, stream>>>(stats, bn2_w, bn2_b, sc2, sh2);
  bn_relu_bf16<<<dim3(NEL / 256), 256, 0, stream>>>(xT0, sc2, sh2);

  // layer 3: conv (raw f32 [b,c,l]) + BN stats
  conv_wmma<3><<<dim3(L_ / 256, C_ / 16, B_), 128, 0, stream>>>(
      Wgen + 2 * wstep, xT0, nullptr, nullptr, h3, stats + 2 * C_);
  bn_finalize<<<1, 256, 0, stream>>>(stats + 2 * C_, bn3_w, bn3_b, sc3, sh3);

  // SE + residual + final relu
  se_mean<<<dim3(B_ * C_), 256, 0, stream>>>(h3, sc3, sh3, sbuf);
  se_gemv<<<dim3(B_), 256, 0, stream>>>(sbuf, se1_w, se1_b, se2_w, se2_b, gate);
  final_fuse<<<dim3(NEL / 256), 256, 0, stream>>>(h3, sc3, sh3, gate, x, out);
}
